// classify_LSTM_63445256896806
// MI455X (gfx1250) — matrix operand order repaired and checked by reading, not compiled, not run on hardware
//
#include <hip/hip_runtime.h>
#include <hip/hip_bf16.h>

// Problem dims (fixed by the reference)
#define B_   64
#define T_   512
#define E_   256
#define Hh_  256
#define H_   512
#define G4_  1024   // 4*Hh
#define K_   16

typedef __attribute__((ext_vector_type(16))) _Float16 v16h;
typedef __attribute__((ext_vector_type(8)))  _Float16 v8h;
typedef __attribute__((ext_vector_type(8)))  float    v8f;

// ---- WMMA fragment loaders (CDNA5 16x16x32 f16 layouts) --------------------
// A (16x32, row-major src, stride ld): lane L<16 -> row L, K = {0..7,16..23};
// lane L>=16 -> row L-16, K = {8..15,24..31}.
__device__ __forceinline__ v16h make_a_frag(const _Float16* base, int ld,
                                            int row0, int k0, int lane) {
  int r = lane & 15, hf = lane >> 4;
  const _Float16* p = base + (size_t)(row0 + r) * ld + k0 + hf * 8;
  v8h lo = *(const v8h*)(p);
  v8h hi = *(const v8h*)(p + 16);
  v16h a;
#pragma unroll
  for (int e = 0; e < 8; ++e) { a[e] = lo[e]; a[e + 8] = hi[e]; }
  return a;
}

// B (32x16) built from row-major weight matrix W[N][ldk]: B[kk][n] = W[n0+n][k0+kk].
// lane = column n (lane&15); half-wave selects K {0..7,16..23} vs {8..15,24..31}.
__device__ __forceinline__ v16h make_b_frag(const _Float16* W, int ld,
                                            int n0, int k0, int lane) {
  int n = lane & 15, hf = lane >> 4;
  const _Float16* p = W + (size_t)(n0 + n) * ld + k0 + hf * 8;
  v8h lo = *(const v8h*)(p);
  v8h hi = *(const v8h*)(p + 16);
  v16h b;
#pragma unroll
  for (int e = 0; e < 8; ++e) { b[e] = lo[e]; b[e + 8] = hi[e]; }
  return b;
}

// ---- Kernel 1: embedding gather + max_norm=1 renorm, f32 -> f16 ------------
__global__ __launch_bounds__(256) void embed_kernel(const int* __restrict__ y,
                                                    const float* __restrict__ table,
                                                    _Float16* __restrict__ emb) {
  __shared__ float red[256];
  int i = blockIdx.x;          // token 0..B*T-1
  int e = threadIdx.x;
  int tok = y[i];
  float v = table[(size_t)tok * E_ + e];
  red[e] = v * v;
  __syncthreads();
  for (int s = 128; s > 0; s >>= 1) {
    if (e < s) red[e] += red[e + s];
    __syncthreads();
  }
  float n = sqrtf(red[0]);
  float scale = fminf(1.0f, 1.0f / fmaxf(n, 1e-7f));
  emb[(size_t)i * E_ + e] = (_Float16)(v * scale);
}

// ---- generic f32 -> f16 convert -------------------------------------------
__global__ void cvt16_kernel(const float* __restrict__ src, _Float16* __restrict__ dst, int n) {
  int i = blockIdx.x * blockDim.x + threadIdx.x;
  if (i < n) dst[i] = (_Float16)src[i];
}

// ---- Kernel 2: xw = emb @ Wi^T + b, both directions, WMMA ------------------
// K-outer, 4 M-tiles per wave: each B fragment loaded once, reused 4x.
// Output layout: xw[dir][t][b][j] (f16) so the LSTM step reads contiguously.
__global__ __launch_bounds__(256) void xw_gemm_kernel(
    const _Float16* __restrict__ emb,
    const _Float16* __restrict__ Wf, const _Float16* __restrict__ Wb,
    const float* __restrict__ bf, const float* __restrict__ bb,
    _Float16* __restrict__ xw) {
  int wave = (blockIdx.x * blockDim.x + threadIdx.x) >> 5;  // 65536 waves total
  int lane = threadIdx.x & 31;
  int dir  = wave >> 15;                                    // 32768 waves/dir
  int tid  = wave & ((1 << 15) - 1);
  int mg = tid >> 6, nt = tid & 63;                         // 512 M-groups x 64 N-tiles
  const _Float16* W  = dir ? Wb : Wf;
  const float* bias  = dir ? bb : bf;
  int m0 = mg * 64, n0 = nt * 16;
  v8f acc[4] = {};
#pragma unroll
  for (int kt = 0; kt < 8; ++kt) {
    v16h b = make_b_frag(W, E_, n0, kt * 32, lane);
#pragma unroll
    for (int m = 0; m < 4; ++m) {
      v16h a = make_a_frag(emb, E_, m0 + m * 16, kt * 32, lane);
      acc[m] = __builtin_amdgcn_wmma_f32_16x16x32_f16(false, a, false, b, (short)0, acc[m], false, false);
    }
  }
  int hf = lane >> 4, col = n0 + (lane & 15);
  float bv = bias[col];
#pragma unroll
  for (int m = 0; m < 4; ++m) {
#pragma unroll
    for (int r = 0; r < 8; ++r) {
      int row = m0 + m * 16 + hf * 8 + r;   // row = b*T + t
      int bb_ = row >> 9, t = row & 511;
      size_t o = (((size_t)dir * T_ + t) * B_ + bb_) * G4_ + col;
      xw[o] = (_Float16)(acc[m][r] + bv);
    }
  }
}

// ---- Kernel 3: persistent LSTM recurrence (one WGP per direction) ----------
// LDS: G (64x1024 f32, 256KB) + h (64x256 f16, 32KB) = 288KB <= 320KB/WGP.
// K-outer per N-tile: each Wh B-fragment loaded once per step, reused for 4 M-tiles.
__global__ __launch_bounds__(1024) void lstm_kernel(
    const _Float16* __restrict__ xw,
    const _Float16* __restrict__ Whf, const _Float16* __restrict__ Whb,
    _Float16* __restrict__ feats) {
  extern __shared__ char smem[];
  float*    Gs = (float*)smem;                              // [64][1024]
  _Float16* hs = (_Float16*)(smem + (size_t)B_ * G4_ * 4);  // [64][256]
  int dir = blockIdx.x;
  const _Float16* Wh = dir ? Whb : Whf;
  int tid = threadIdx.x, wave = tid >> 5, lane = tid & 31;

  for (int i = tid; i < B_ * Hh_; i += 1024) hs[i] = (_Float16)0.0f;
  float c[16];
#pragma unroll
  for (int v = 0; v < 16; ++v) c[v] = 0.0f;
  __syncthreads();

  int n0 = wave * 32;  // this wave owns N-tiles n0, n0+16 across all 4 M-tiles
  for (int step = 0; step < T_; ++step) {
    int t = dir ? (T_ - 1 - step) : step;
    // G = h @ Wh^T  (64x256 @ 256x1024) via WMMA, h in LDS, Wh from L2
#pragma unroll
    for (int ni = 0; ni < 2; ++ni) {
      int n = n0 + ni * 16;
      v8f acc[4] = {};
#pragma unroll
      for (int kt = 0; kt < 8; ++kt) {
        v16h b = make_b_frag(Wh, Hh_, n, kt * 32, lane);
#pragma unroll
        for (int m = 0; m < 4; ++m) {
          v16h a = make_a_frag(hs, Hh_, m * 16, kt * 32, lane);
          acc[m] = __builtin_amdgcn_wmma_f32_16x16x32_f16(false, a, false, b, (short)0, acc[m], false, false);
        }
      }
      int hf = lane >> 4, col = n + (lane & 15);
#pragma unroll
      for (int m = 0; m < 4; ++m)
#pragma unroll
        for (int r = 0; r < 8; ++r)
          Gs[(size_t)(m * 16 + hf * 8 + r) * G4_ + col] = acc[m][r];
    }
    __syncthreads();  // all hs reads + Gs writes done

    // Gate math: each thread owns 16 fixed (b,hh) cells; c lives in registers.
    const _Float16* xwt = xw + (((size_t)dir * T_ + t) * B_) * G4_;
#pragma unroll
    for (int v = 0; v < 16; ++v) {
      int idx = v * 1024 + tid;
      int b = idx >> 8, hh = idx & 255;
      const float*    Gb = Gs  + (size_t)b * G4_;
      const _Float16* xb = xwt + (size_t)b * G4_;
      float gi = Gb[hh]        + (float)xb[hh];
      float gf = Gb[256 + hh]  + (float)xb[256 + hh];
      float gg = Gb[512 + hh]  + (float)xb[512 + hh];
      float go = Gb[768 + hh]  + (float)xb[768 + hh];
      gi = 1.0f / (1.0f + expf(-gi));
      gf = 1.0f / (1.0f + expf(-gf));
      gg = tanhf(gg);
      go = 1.0f / (1.0f + expf(-go));
      c[v] = gf * c[v] + gi * gg;
      float h = go * tanhf(c[v]);
      hs[(size_t)b * Hh_ + hh] = (_Float16)h;
      feats[((size_t)b * T_ + t) * H_ + dir * Hh_ + hh] = (_Float16)h;
    }
    __syncthreads();  // hs fully updated before next step's WMMA
  }
}

// ---- Kernel 4: emissions = feats @ lin_W^T + lin_b (N=16 = one tile) -------
// 4 M-tiles per wave, K-outer: lin_W fragments reused 4x.
__global__ __launch_bounds__(256) void emis_kernel(
    const _Float16* __restrict__ feats, const _Float16* __restrict__ linW,
    const float* __restrict__ linb, float* __restrict__ emis) {
  int wave = (blockIdx.x * blockDim.x + threadIdx.x) >> 5;  // 512 waves
  int lane = threadIdx.x & 31;
  int m0 = wave * 64;
  v8f acc[4] = {};
#pragma unroll
  for (int kt = 0; kt < 16; ++kt) {
    v16h b = make_b_frag(linW, H_, 0, kt * 32, lane);
#pragma unroll
    for (int m = 0; m < 4; ++m) {
      v16h a = make_a_frag(feats, H_, m0 + m * 16, kt * 32, lane);
      acc[m] = __builtin_amdgcn_wmma_f32_16x16x32_f16(false, a, false, b, (short)0, acc[m], false, false);
    }
  }
  int hf = lane >> 4, col = lane & 15;
  float bv = linb[col];
#pragma unroll
  for (int m = 0; m < 4; ++m)
#pragma unroll
    for (int r = 0; r < 8; ++r) {
      int row = m0 + m * 16 + hf * 8 + r;
      emis[(size_t)row * K_ + col] = acc[m][r] + bv;
    }
}

// ---- Kernel 5: CRF Viterbi, one wave per batch row -------------------------
__global__ __launch_bounds__(32) void viterbi_kernel(
    const float* __restrict__ emis, const unsigned char* __restrict__ mask,
    const float* __restrict__ start_t, const float* __restrict__ trans,
    const float* __restrict__ end_t, unsigned char* __restrict__ hist,
    float* __restrict__ out) {
  int b = blockIdx.x;
  int k = threadIdx.x;
  __shared__ float sc[K_];
  __shared__ float tr[K_ * K_];
  if (k < K_)
    for (int i = k; i < K_ * K_; i += K_) tr[i] = trans[i];
  float score = 0.0f;
  if (k < K_) score = start_t[k] + emis[(size_t)b * T_ * K_ + k];
  __syncthreads();
  for (int t = 1; t < T_; ++t) {
    if (k < K_) sc[k] = score;
    __syncthreads();
    if (k < K_) {
      float best = -3.4e38f; int bi = 0;
      for (int kf = 0; kf < K_; ++kf) {
        float v = sc[kf] + tr[kf * K_ + k];
        if (v > best) { best = v; bi = kf; }   // first-max tie-break like argmax
      }
      float emit = emis[((size_t)b * T_ + t) * K_ + k];
      hist[((size_t)(t - 1) * B_ + b) * K_ + k] = (unsigned char)bi;
      score = mask[b * T_ + t] ? (best + emit) : score;
    }
    __syncthreads();
  }
  if (k < K_) sc[k] = score + end_t[k];
  __syncthreads();
  if (k == 0) {
    float best = sc[0]; int cur = 0;
    for (int j = 1; j < K_; ++j) if (sc[j] > best) { best = sc[j]; cur = j; }
    out[(size_t)b * T_ + (T_ - 1)] = (float)cur;
    for (int t = T_ - 2; t >= 0; --t) {
      int prev = hist[((size_t)t * B_ + b) * K_ + cur];
      if (mask[b * T_ + t + 1]) cur = prev;
      out[(size_t)b * T_ + t] = (float)cur;
    }
  }
}

// ---------------------------------------------------------------------------
extern "C" void kernel_launch(void* const* d_in, const int* in_sizes, int n_in,
                              void* d_out, int out_size, void* d_ws, size_t ws_size,
                              hipStream_t stream) {
  (void)in_sizes; (void)n_in; (void)out_size; (void)ws_size;
  const int*           y       = (const int*)d_in[0];
  const unsigned char* mask    = (const unsigned char*)d_in[1];
  const float*         table   = (const float*)d_in[2];
  const float*         Wi_f    = (const float*)d_in[3];
  const float*         Wh_f    = (const float*)d_in[4];
  const float*         b_f     = (const float*)d_in[5];
  const float*         Wi_b    = (const float*)d_in[6];
  const float*         Wh_b    = (const float*)d_in[7];
  const float*         b_b     = (const float*)d_in[8];
  const float*         lin_W   = (const float*)d_in[9];
  const float*         lin_b   = (const float*)d_in[10];
  const float*         start_t = (const float*)d_in[11];
  const float*         trans   = (const float*)d_in[12];
  const float*         end_t   = (const float*)d_in[13];
  float* out = (float*)d_out;

  // workspace carve-up (256B aligned)
  char* ws = (char*)d_ws;
  size_t off = 0;
  auto carve = [&](size_t bytes) {
    size_t o = off; off += (bytes + 255) & ~(size_t)255; return o;
  };
  _Float16* emb16   = (_Float16*)(ws + carve((size_t)B_ * T_ * E_ * 2));
  _Float16* wif16   = (_Float16*)(ws + carve((size_t)G4_ * E_ * 2));
  _Float16* wib16   = (_Float16*)(ws + carve((size_t)G4_ * E_ * 2));
  _Float16* whf16   = (_Float16*)(ws + carve((size_t)G4_ * Hh_ * 2));
  _Float16* whb16   = (_Float16*)(ws + carve((size_t)G4_ * Hh_ * 2));
  _Float16* lw16    = (_Float16*)(ws + carve((size_t)K_ * H_ * 2));
  _Float16* xw16    = (_Float16*)(ws + carve((size_t)2 * T_ * B_ * G4_ * 2));
  _Float16* feats16 = (_Float16*)(ws + carve((size_t)B_ * T_ * H_ * 2));
  float*    emis    = (float*)   (ws + carve((size_t)B_ * T_ * K_ * 4));
  unsigned char* hist = (unsigned char*)(ws + carve((size_t)(T_ - 1) * B_ * K_));

  const int lstm_smem = B_ * G4_ * 4 + B_ * Hh_ * 2;  // 288KB
  hipFuncSetAttribute((const void*)lstm_kernel,
                      hipFuncAttributeMaxDynamicSharedMemorySize, lstm_smem);

  embed_kernel<<<B_ * T_, 256, 0, stream>>>(y, table, emb16);
  cvt16_kernel<<<(G4_ * E_ + 255) / 256, 256, 0, stream>>>(Wi_f, wif16, G4_ * E_);
  cvt16_kernel<<<(G4_ * E_ + 255) / 256, 256, 0, stream>>>(Wi_b, wib16, G4_ * E_);
  cvt16_kernel<<<(G4_ * Hh_ + 255) / 256, 256, 0, stream>>>(Wh_f, whf16, G4_ * Hh_);
  cvt16_kernel<<<(G4_ * Hh_ + 255) / 256, 256, 0, stream>>>(Wh_b, whb16, G4_ * Hh_);
  cvt16_kernel<<<(K_ * H_ + 255) / 256, 256, 0, stream>>>(lin_W, lw16, K_ * H_);

  // 65536 waves: 2 dirs x 512 M-groups x 64 N-tiles, 8 waves/block
  xw_gemm_kernel<<<8192, 256, 0, stream>>>(emb16, wif16, wib16, b_f, b_b, xw16);
  lstm_kernel<<<2, 1024, lstm_smem, stream>>>(xw16, whf16, whb16, feats16);
  emis_kernel<<<64, 256, 0, stream>>>(feats16, lw16, lin_b, emis);
  viterbi_kernel<<<B_, 32, 0, stream>>>(emis, mask, start_t, trans, end_t, hist, out);
}
